// RRLSTM_28535762715372
// MI455X (gfx1250) — matrix operand order repaired and checked by reading, compiled, not run on hardware
//
#include <hip/hip_runtime.h>
#include <hip/hip_bf16.h>

// Problem constants (match reference).
#define Bsz   128
#define Tt    512
#define T1    513   // T+1
#define Sdim  64
#define Adim  8
#define Udim  512
#define INP   72
#define KPAD  96    // 72 padded to 3 x 32 for WMMA K
#define HSTR  520   // LDS row stride in halfs (512 + 8 pad, conflict-free b128)
#define NWAVE 16    // waves per block
#define NT    2     // N-tiles (of 16) per wave -> 16*2*16 = 512 units

typedef __attribute__((ext_vector_type(16))) __bf16 v16bf;
typedef __attribute__((ext_vector_type(8)))  float  v8f;
typedef unsigned short u16;

struct alignas(16) U128 { unsigned int a, b, c, d; };
struct alignas(16) F4   { float x, y, z, w; };

union Frag16 {
    v16bf v;
    u16   u[16];
    U128  q[2];
};

__device__ __forceinline__ u16 f2bf(float f) {
    unsigned u = __float_as_uint(f);
    unsigned r = u + 0x7FFFu + ((u >> 16) & 1u);   // round-to-nearest-even
    return (u16)(r >> 16);
}

__device__ __forceinline__ float sigmoidf_fast(float x) {
    return 1.0f / (1.0f + __expf(-x));
}

__device__ __forceinline__ float tanhf_fast(float x) {
    // tanh(x) = 1 - 2/(exp(2x)+1)
    float e = __expf(2.0f * x);
    return 1.0f - 2.0f / (e + 1.0f);
}

// ---------------------------------------------------------------------------
// Kernel 1: transpose+convert weights to bf16 [N][K] layout in workspace.
// ---------------------------------------------------------------------------
__global__ __launch_bounds__(256) void repack_kernel(
    const float* __restrict__ W_ci, const float* __restrict__ W_og,
    const float* __restrict__ W_ig,
    u16* __restrict__ wig_t, u16* __restrict__ wci_t, u16* __restrict__ wog_t)
{
    int idx = blockIdx.x * 256 + threadIdx.x;
    if (idx < Udim * Udim) {
        int n = idx >> 9, k = idx & 511;
        wig_t[idx] = f2bf(W_ig[(size_t)k * Udim + n]);
    }
    if (idx < Udim * KPAD) {
        int n = idx / KPAD, k = idx % KPAD;
        float vc = (k < INP) ? W_ci[(size_t)k * Udim + n] : 0.0f;
        float vo = (k < INP) ? W_og[(size_t)k * Udim + n] : 0.0f;
        wci_t[idx] = f2bf(vc);
        wog_t[idx] = f2bf(vo);
    }
}

// ---------------------------------------------------------------------------
// Kernel 2: fused recurrent scan. One block = 16 batch rows, 16 waves split
// the 512 output units (32 each). h shared via LDS, 2 barriers / step.
// Weights stream from L2 each step (working set ~0.7 MB << 192 MB L2).
// ---------------------------------------------------------------------------
__global__ __launch_bounds__(NWAVE * 32) void rlstm_kernel(
    const float* __restrict__ states,   // [B, T1, S]
    const int*   __restrict__ actions,  // [B, T]
    const float* __restrict__ b_ci, const float* __restrict__ b_ig,
    const float* __restrict__ b_og,
    const float* __restrict__ W_lin, const float* __restrict__ b_lin,
    const u16* wig_t,   // [512][512] bf16   (no __restrict: keep loads in-loop)
    const u16* wci_t,   // [512][96]  bf16
    const u16* wog_t,   // [512][96]  bf16
    float* __restrict__ out)            // [B, T1]
{
    __shared__ alignas(16) u16 hlds[16 * HSTR];  // h tile, bf16, row-major
    __shared__ float part[2][16];                // head partials, double-buffered

    const int tid   = threadIdx.x;
    const int lane  = tid & 31;
    const int wave  = tid >> 5;
    const int m     = lane & 15;      // row within 16-row tile / N within 16
    const int hi    = lane >> 4;      // K-half selector (0/1)
    const int nbase = wave * (NT * 16);
    const int bm    = blockIdx.x * 16;

    for (int i = tid; i < 16 * HSTR; i += NWAVE * 32) hlds[i] = 0;
    if (tid < 32) part[tid >> 4][tid & 15] = 0.0f;
    __syncthreads();

    // Per-lane constants (one N column per (ntile, lane%16) position).
    float bciv[NT], bigv[NT], bogv[NT], wlv[NT];
#pragma unroll
    for (int nt = 0; nt < NT; ++nt) {
        int n = nbase + nt * 16 + m;
        bciv[nt] = b_ci[n]; bigv[nt] = b_ig[n];
        bogv[nt] = b_og[n]; wlv[nt]  = W_lin[n];
    }
    const float blin = b_lin[0];

    v8f c[NT];
#pragma unroll
    for (int nt = 0; nt < NT; ++nt)
#pragma unroll
        for (int i = 0; i < 8; ++i) c[nt][i] = 0.0f;

    const int    brow = bm + m;
    const float* srow = states  + (size_t)brow * T1 * Sdim;
    const int*   arow = actions + (size_t)brow * Tt;

    // Per-wave weight bases (N fixed per lane for whole kernel).
    const int n0 = nbase + m;          // nt = 0 column
    const int n1 = nbase + 16 + m;     // nt = 1 column

    for (int t = 0; t < T1; ++t) {
        // Defeat LICM: without this, the t-invariant weight fragments get
        // hoisted out of the loop and spilled to scratch (observed round 1).
        asm volatile("" ::: "memory");

        // ---------- phase A: ig accumulator = h_prev @ W_ig ----------
        v8f cig0, cig1;
#pragma unroll
        for (int i = 0; i < 8; ++i) { cig0[i] = 0.0f; cig1[i] = 0.0f; }

        for (int kt = 0; kt < 16; ++kt) {
            const int  p0  = kt * 32 + hi * 8;
            const int  klo = kt * 32 + hi * 8;
            const u16* hp  = &hlds[m * HSTR + p0];
            const u16* w0  = wig_t + (size_t)n0 * Udim + klo;
            const u16* w1  = wig_t + (size_t)n1 * Udim + klo;
            Frag16 a, b0, b1;
            a.q[0]  = *(const U128*)hp;
            a.q[1]  = *(const U128*)(hp + 16);
            b0.q[0] = *(const U128*)w0;  b0.q[1] = *(const U128*)(w0 + 16);
            b1.q[0] = *(const U128*)w1;  b1.q[1] = *(const U128*)(w1 + 16);
            cig0 = __builtin_amdgcn_wmma_f32_16x16x32_bf16(
                false, a.v, false, b0.v, (short)0, cig0, false, false);
            cig1 = __builtin_amdgcn_wmma_f32_16x16x32_bf16(
                false, a.v, false, b1.v, (short)0, cig1, false, false);
        }

        // ---------- phase B: feedforward ci/og = x_t @ W ----------
        v8f cci0, cci1, cog0, cog1;
#pragma unroll
        for (int i = 0; i < 8; ++i) {
            cci0[i] = 0.0f; cci1[i] = 0.0f; cog0[i] = 0.0f; cog1[i] = 0.0f;
        }

        const float* xp  = srow + (size_t)t * Sdim;
        const int    act = (t < Tt) ? arow[t] : -1;

#pragma unroll
        for (int kt = 0; kt < 3; ++kt) {
            Frag16 a;
            const int p0 = kt * 32 + hi * 8;
            if (kt < 2) {
                // pure state features, 8 + 8 consecutive f32 -> bf16
                F4 f0 = *(const F4*)(xp + p0);
                F4 f1 = *(const F4*)(xp + p0 + 4);
                F4 f2 = *(const F4*)(xp + p0 + 16);
                F4 f3 = *(const F4*)(xp + p0 + 20);
                a.u[0]  = f2bf(f0.x); a.u[1]  = f2bf(f0.y);
                a.u[2]  = f2bf(f0.z); a.u[3]  = f2bf(f0.w);
                a.u[4]  = f2bf(f1.x); a.u[5]  = f2bf(f1.y);
                a.u[6]  = f2bf(f1.z); a.u[7]  = f2bf(f1.w);
                a.u[8]  = f2bf(f2.x); a.u[9]  = f2bf(f2.y);
                a.u[10] = f2bf(f2.z); a.u[11] = f2bf(f2.w);
                a.u[12] = f2bf(f3.x); a.u[13] = f2bf(f3.y);
                a.u[14] = f2bf(f3.z); a.u[15] = f2bf(f3.w);
            } else {
                // one-hot action region (k 64..71) + zero pad (k 72..95)
#pragma unroll
                for (int e = 0; e < 8; ++e) {
                    int k0 = p0 + e;
                    a.u[e]     = (k0 < 72 && (k0 - 64) == act) ? (u16)0x3F80u : (u16)0;
                    a.u[8 + e] = 0;   // k0+16 >= 80 always pad
                }
            }
            const int  klo = kt * 32 + hi * 8;
            const u16* wc0 = wci_t + (size_t)n0 * KPAD + klo;
            const u16* wc1 = wci_t + (size_t)n1 * KPAD + klo;
            const u16* wo0 = wog_t + (size_t)n0 * KPAD + klo;
            const u16* wo1 = wog_t + (size_t)n1 * KPAD + klo;
            Frag16 bc0, bc1, bo0, bo1;
            bc0.q[0] = *(const U128*)wc0; bc0.q[1] = *(const U128*)(wc0 + 16);
            bc1.q[0] = *(const U128*)wc1; bc1.q[1] = *(const U128*)(wc1 + 16);
            bo0.q[0] = *(const U128*)wo0; bo0.q[1] = *(const U128*)(wo0 + 16);
            bo1.q[0] = *(const U128*)wo1; bo1.q[1] = *(const U128*)(wo1 + 16);
            cci0 = __builtin_amdgcn_wmma_f32_16x16x32_bf16(
                false, a.v, false, bc0.v, (short)0, cci0, false, false);
            cci1 = __builtin_amdgcn_wmma_f32_16x16x32_bf16(
                false, a.v, false, bc1.v, (short)0, cci1, false, false);
            cog0 = __builtin_amdgcn_wmma_f32_16x16x32_bf16(
                false, a.v, false, bo0.v, (short)0, cog0, false, false);
            cog1 = __builtin_amdgcn_wmma_f32_16x16x32_bf16(
                false, a.v, false, bo1.v, (short)0, cog1, false, false);
        }

        // ---------- elementwise cell update + head partials ----------
        float hsum[8];
#pragma unroll
        for (int i = 0; i < 8; ++i) hsum[i] = 0.0f;
        u16 hb[NT][8];
#pragma unroll
        for (int nt = 0; nt < NT; ++nt) {
            const v8f& gi = (nt == 0) ? cig0 : cig1;
            const v8f& gc = (nt == 0) ? cci0 : cci1;
            const v8f& go = (nt == 0) ? cog0 : cog1;
#pragma unroll
            for (int i = 0; i < 8; ++i) {
                float ig  = sigmoidf_fast(gi[i] + bigv[nt]);
                float civ = tanhf_fast(gc[i] + bciv[nt]);
                float ogv = sigmoidf_fast(go[i] + bogv[nt]);
                float cv  = c[nt][i] + civ * ig;   // no forget gate
                c[nt][i]  = cv;
                float hv  = cv * ogv;
                hsum[i]  += hv * wlv[nt];
                hb[nt][i] = f2bf(hv);
            }
        }

        __syncthreads();   // all waves done reading hlds (phase A)

        // store new h tile (C-fragment layout: row = i + 8*hi, col = n)
#pragma unroll
        for (int nt = 0; nt < NT; ++nt) {
            const int n = nbase + nt * 16 + m;
#pragma unroll
            for (int i = 0; i < 8; ++i)
                hlds[(i + 8 * hi) * HSTR + n] = hb[nt][i];
        }

        // head: reduce over this wave's 16 N-lanes, then across waves via LDS
#pragma unroll
        for (int mask = 1; mask < 16; mask <<= 1)
#pragma unroll
            for (int i = 0; i < 8; ++i)
                hsum[i] += __shfl_xor(hsum[i], mask);
        if (m == 0) {
#pragma unroll
            for (int i = 0; i < 8; ++i)
                atomicAdd(&part[t & 1][i + 8 * hi], hsum[i]);
        }

        __syncthreads();   // hlds + part[t&1] complete

        if (tid < 16) {
            out[(size_t)(bm + tid) * T1 + t] = part[t & 1][tid] + blin;
            part[t & 1][tid] = 0.0f;   // recycle buffer for step t+2
        }
    }
}

// ---------------------------------------------------------------------------
extern "C" void kernel_launch(void* const* d_in, const int* in_sizes, int n_in,
                              void* d_out, int out_size, void* d_ws, size_t ws_size,
                              hipStream_t stream) {
    const float* states  = (const float*)d_in[0];
    const int*   actions = (const int*)  d_in[1];
    const float* W_ci    = (const float*)d_in[2];
    const float* b_ci    = (const float*)d_in[3];
    const float* W_ig    = (const float*)d_in[4];
    const float* b_ig    = (const float*)d_in[5];
    const float* W_og    = (const float*)d_in[6];
    const float* b_og    = (const float*)d_in[7];
    const float* W_lin   = (const float*)d_in[8];
    const float* b_lin   = (const float*)d_in[9];

    // Workspace layout (bf16 transposed weights), ~0.7 MB total.
    u16* wig_t = (u16*)d_ws;                 // 512*512
    u16* wci_t = wig_t + Udim * Udim;        // 512*96
    u16* wog_t = wci_t + Udim * KPAD;        // 512*96

    repack_kernel<<<(Udim * Udim + 255) / 256, 256, 0, stream>>>(
        W_ci, W_og, W_ig, wig_t, wci_t, wog_t);

    rlstm_kernel<<<Bsz / 16, NWAVE * 32, 0, stream>>>(
        states, actions, b_ci, b_ig, b_og, W_lin, b_lin,
        wig_t, wci_t, wog_t, (float*)d_out);
}
